// SimpleLSTMModel_66932770341481
// MI455X (gfx1250) — matrix operand order repaired and checked by reading, not compiled, not run on hardware
//
#include <hip/hip_runtime.h>
#include <hip/hip_bf16.h>

typedef __attribute__((ext_vector_type(16))) __bf16 bf16x16;
typedef __attribute__((ext_vector_type(8)))  float  f32x8;

#define T_STEPS 162
#define HID 64
#define NG 256            // 4*HID gate width
#define BLOCK 128         // 4 waves per workgroup
#define WAVES 4
#define TILE_B 64         // 16 batch rows per wave

__device__ __forceinline__ unsigned short f2bf(float f) {
    unsigned int u = __builtin_bit_cast(unsigned int, f);
    u += 0x7FFFu + ((u >> 16) & 1u);           // round-to-nearest-even
    return (unsigned short)(u >> 16);
}
__device__ __forceinline__ float bf2f(unsigned short s) {
    unsigned int u = ((unsigned int)s) << 16;
    return __builtin_bit_cast(float, u);
}

// CDNA5 has a hardware V_TANH_F32 (single TRANS op); fall back to exp form.
#if __has_builtin(__builtin_amdgcn_tanhf)
__device__ __forceinline__ float tanh_f(float x) { return __builtin_amdgcn_tanhf(x); }
#elif __has_builtin(__builtin_amdgcn_tanh_f32)
__device__ __forceinline__ float tanh_f(float x) { return __builtin_amdgcn_tanh_f32(x); }
#else
__device__ __forceinline__ float tanh_f(float x) {
    return 2.0f * __builtin_amdgcn_rcpf(1.0f + __expf(-2.0f * x)) - 1.0f;
}
#endif
__device__ __forceinline__ float sigmoid_f(float x) {
    return __builtin_fmaf(0.5f, tanh_f(0.5f * x), 0.5f);
}

union ABCast { uint4 u[2]; bf16x16 v; };

// A fragment (16x32 bf16): lane holds row M=lane%16; lanes 0-15 K=c..c+7 & c+16..c+23
// with c = chunk*32; lanes 16-31 same +8.  Source scratch is row-major [16][64] bf16.
__device__ __forceinline__ bf16x16 load_a_frag(const unsigned short* scr, int lane, int chunk) {
    int base = (lane & 15) * HID + chunk * 32 + ((lane >> 4) << 3);
    ABCast c;
    c.u[0] = *(const uint4*)(scr + base);        // K = c0 .. c0+7
    c.u[1] = *(const uint4*)(scr + base + 16);   // K = c0+16 .. c0+23
    return c.v;
}
// B fragment (32x16 bf16): lane holds column N = ntile*16 + lane%16;
// lanes 0-15 hold K = c..c+7 & c+16..c+23, lanes 16-31 same +8 (c = chunk*32).
// W stored row-major [256][64] so B[k][n] = W[n][k] -> two contiguous runs of 8 bf16 per lane.
__device__ __forceinline__ bf16x16 load_b_frag(const unsigned short* w, int lane, int ntile, int chunk) {
    int n  = ntile * 16 + (lane & 15);
    int k0 = chunk * 32 + ((lane >> 4) << 3);
    const unsigned short* p = w + n * HID + k0;
    ABCast c;
    c.u[0] = *(const uint4*)(p);
    c.u[1] = *(const uint4*)(p + 16);
    return c.v;
}
__device__ __forceinline__ f32x8 wmma_bf16(bf16x16 a, bf16x16 b, f32x8 c) {
    return __builtin_amdgcn_wmma_f32_16x16x32_bf16(false, a, false, b, (short)0, c, false, false);
}

extern "C" __global__ void __launch_bounds__(BLOCK, 1) lstm2_wmma_kernel(
    const float* __restrict__ x,      // [B, T]
    const float* __restrict__ wih0,   // [256, 1]
    const float* __restrict__ whh0,   // [256, 64]
    const float* __restrict__ bih0,
    const float* __restrict__ bhh0,
    const float* __restrict__ wih1,   // [256, 64]
    const float* __restrict__ whh1,   // [256, 64]
    const float* __restrict__ bih1,
    const float* __restrict__ bhh1,
    const float* __restrict__ fcw,    // [64]
    const float* __restrict__ fcb,    // [1]
    float* __restrict__ out, int Bsz)
{
    extern __shared__ char smem[];
    unsigned short* sWhh0 = (unsigned short*)smem;           // 256*64 bf16 (staging only)
    unsigned short* sWih1 = sWhh0 + NG * HID;                // resident all run
    unsigned short* sWhh1 = sWih1 + NG * HID;                // staging only
    float* sGB0  = (float*)(sWhh1 + NG * HID);               // bih0+bhh0
    float* sGB1  = sGB0 + NG;                                // bih1+bhh1
    float* sWih0 = sGB1 + NG;                                // wih0 column
    unsigned short* sH = (unsigned short*)(sWih0 + NG);      // WAVES*2*16*64 bf16

    const int tid = threadIdx.x;
    // ---- one-time weight stage: fp32 global -> bf16 LDS ----
    for (int i = tid; i < NG * HID; i += BLOCK) {
        sWhh0[i] = f2bf(whh0[i]);
        sWih1[i] = f2bf(wih1[i]);
        sWhh1[i] = f2bf(whh1[i]);
    }
    for (int i = tid; i < NG; i += BLOCK) {
        sGB0[i]  = bih0[i] + bhh0[i];
        sGB1[i]  = bih1[i] + bhh1[i];
        sWih0[i] = wih0[i];
    }
    __syncthreads();   // only barrier in the kernel

    const int wave = tid >> 5;
    const int lane = tid & 31;
    const int ln   = lane & 15;
    const int hi   = lane >> 4;
    unsigned short* h0scr = sH + wave * (2 * 16 * HID);      // per-wave private
    unsigned short* h1scr = h0scr + 16 * HID;
    const int rowbase = blockIdx.x * TILE_B + wave * 16;

    // zero h scratch (h(-1) = 0); per-wave region, same-wave LDS is in-order
    for (int i = lane; i < 2 * 16 * HID; i += 32) h0scr[i] = 0;

    // ---- pin the two recurrent weight matrices' B-fragments in VGPRs ----
    // 16 tiles x 2 K-chunks x 8 VGPRs x 2 matrices = 512 VGPRs, live across the T loop.
    bf16x16 cW0[32], cW1[32];
    #pragma unroll
    for (int tl = 0; tl < 16; ++tl) {
        cW0[tl * 2 + 0] = load_b_frag(sWhh0, lane, tl, 0);
        cW0[tl * 2 + 1] = load_b_frag(sWhh0, lane, tl, 1);
        cW1[tl * 2 + 0] = load_b_frag(sWhh1, lane, tl, 0);
        cW1[tl * 2 + 1] = load_b_frag(sWhh1, lane, tl, 1);
    }

    // hoist per-lane gate constants (N = g*64 + jb*16 + ln) out of the T loop
    float wv0[16], bv0[16], bv1[16];
    #pragma unroll
    for (int g = 0; g < 4; ++g)
        #pragma unroll
        for (int jb = 0; jb < 4; ++jb) {
            int n = g * HID + jb * 16 + ln;
            wv0[g * 4 + jb] = sWih0[n];
            bv0[g * 4 + jb] = sGB0[n];
            bv1[g * 4 + jb] = sGB1[n];
        }

    f32x8 c0[4] = {}, c1[4] = {};    // cell state lives in VGPRs for all 162 steps
    const float* xrow = x + (rowbase + 8 * hi) * T_STEPS;

    #pragma unroll 1
    for (int t = 0; t < T_STEPS; ++t) {
        // x for this step; D-layout element r maps to batch row M = r + 8*hi
        float xv[8];
        #pragma unroll
        for (int r = 0; r < 8; ++r) xv[r] = xrow[r * T_STEPS + t];

        // ================= layer 0 =================
        bf16x16 a0lo = load_a_frag(h0scr, lane, 0);   // h0[t-1]
        bf16x16 a0hi = load_a_frag(h0scr, lane, 1);

        #pragma unroll
        for (int jb = 0; jb < 4; ++jb) {
            f32x8 acc[4];
            #pragma unroll
            for (int g = 0; g < 4; ++g) {
                int tl = g * 4 + jb;
                #pragma unroll
                for (int r = 0; r < 8; ++r)
                    acc[g][r] = xv[r] * wv0[tl] + bv0[tl];   // rank-1 input + bias
                acc[g] = wmma_bf16(a0lo, cW0[tl * 2 + 0], acc[g]);
                acc[g] = wmma_bf16(a0hi, cW0[tl * 2 + 1], acc[g]);
            }
            #pragma unroll
            for (int r = 0; r < 8; ++r) {
                float ig = sigmoid_f(acc[0][r]);
                float fg = sigmoid_f(acc[1][r]);
                float gg = tanh_f(acc[2][r]);
                float og = sigmoid_f(acc[3][r]);
                float c  = fg * c0[jb][r] + ig * gg;
                c0[jb][r] = c;
                h0scr[(r + 8 * hi) * HID + jb * 16 + ln] = f2bf(og * tanh_f(c));
            }
        }

        // ================= layer 1 =================
        bf16x16 a1ilo = load_a_frag(h0scr, lane, 0);  // h0[t]  (just written; LDS in-order)
        bf16x16 a1ihi = load_a_frag(h0scr, lane, 1);
        bf16x16 a1rlo = load_a_frag(h1scr, lane, 0);  // h1[t-1]
        bf16x16 a1rhi = load_a_frag(h1scr, lane, 1);

        #pragma unroll
        for (int jb = 0; jb < 4; ++jb) {
            f32x8 acc[4];
            // input contribution (Wih1, streamed from LDS)
            {
                bf16x16 b[8];
                #pragma unroll
                for (int g = 0; g < 4; ++g) {
                    b[g * 2 + 0] = load_b_frag(sWih1, lane, g * 4 + jb, 0);
                    b[g * 2 + 1] = load_b_frag(sWih1, lane, g * 4 + jb, 1);
                }
                #pragma unroll
                for (int g = 0; g < 4; ++g) {
                    #pragma unroll
                    for (int r = 0; r < 8; ++r) acc[g][r] = bv1[g * 4 + jb];
                    acc[g] = wmma_bf16(a1ilo, b[g * 2 + 0], acc[g]);
                    acc[g] = wmma_bf16(a1ihi, b[g * 2 + 1], acc[g]);
                }
            }
            // recurrent contribution (Whh1, pinned in VGPRs)
            #pragma unroll
            for (int g = 0; g < 4; ++g) {
                int tl = g * 4 + jb;
                acc[g] = wmma_bf16(a1rlo, cW1[tl * 2 + 0], acc[g]);
                acc[g] = wmma_bf16(a1rhi, cW1[tl * 2 + 1], acc[g]);
            }
            #pragma unroll
            for (int r = 0; r < 8; ++r) {
                float ig = sigmoid_f(acc[0][r]);
                float fg = sigmoid_f(acc[1][r]);
                float gg = tanh_f(acc[2][r]);
                float og = sigmoid_f(acc[3][r]);
                float c  = fg * c1[jb][r] + ig * gg;
                c1[jb][r] = c;
                h1scr[(r + 8 * hi) * HID + jb * 16 + ln] = f2bf(og * tanh_f(c));
            }
        }
    }

    // ---- final FC: out[b] = h1_final[b,:] . fcw + fcb ----
    if (hi == 0) {
        float acc = 0.f;
        #pragma unroll
        for (int k = 0; k < HID; ++k)
            acc += bf2f(h1scr[ln * HID + k]) * fcw[k];
        int row = rowbase + ln;
        if (row < Bsz) out[row] = acc + fcb[0];
    }
}

extern "C" void kernel_launch(void* const* d_in, const int* in_sizes, int n_in,
                              void* d_out, int out_size, void* d_ws, size_t ws_size,
                              hipStream_t stream) {
    (void)n_in; (void)d_ws; (void)ws_size; (void)out_size;
    const float* x    = (const float*)d_in[0];
    const float* wih0 = (const float*)d_in[1];
    const float* whh0 = (const float*)d_in[2];
    const float* bih0 = (const float*)d_in[3];
    const float* bhh0 = (const float*)d_in[4];
    const float* wih1 = (const float*)d_in[5];
    const float* whh1 = (const float*)d_in[6];
    const float* bih1 = (const float*)d_in[7];
    const float* bhh1 = (const float*)d_in[8];
    const float* fcw  = (const float*)d_in[9];
    const float* fcb  = (const float*)d_in[10];
    float* out = (float*)d_out;

    const int B = in_sizes[0] / T_STEPS;
    const int grid = (B + TILE_B - 1) / TILE_B;
    const size_t smem = (size_t)(3 * NG * HID * 2)          // 3 bf16 weight matrices
                      + (size_t)(3 * NG * 4)                // gb0, gb1, wih0 (fp32)
                      + (size_t)(WAVES * 2 * 16 * HID * 2); // per-wave h scratch
    lstm2_wmma_kernel<<<grid, BLOCK, smem, stream>>>(
        x, wih0, whh0, bih0, bhh0, wih1, whh1, bih1, bhh1, fcw, fcb, out, B);
}
